// StructuredLatentSpace_85779086836169
// MI455X (gfx1250) — compile-verified, matrix operand order checked by reading, not run on hardware
//
#include <hip/hip_runtime.h>
#include <hip/hip_bf16.h>

typedef __attribute__((ext_vector_type(16))) _Float16 v16h;
typedef __attribute__((ext_vector_type(8)))  _Float16 v8h;
typedef __attribute__((ext_vector_type(8)))  float    v8f;

#define K_DIM   128
#define T_DIM   1024
#define ATOMS   256
#define OUT_ROWS 1280   // 256 k + 256 theta + 768 logits

// ---------------------------------------------------------------------------
// Prep kernel 1: W (1280x128 f32, row-major) -> f16 row-major
// ---------------------------------------------------------------------------
__global__ __launch_bounds__(256)
void convert_w_kernel(const float* __restrict__ W, _Float16* __restrict__ Wh) {
    int i = blockIdx.x * 256 + threadIdx.x;   // grid covers 1280*128 exactly
    Wh[i] = (_Float16)W[i];
}

// ---------------------------------------------------------------------------
// Prep kernel 2: h[b][k][t] f32 -> Ht[b][t][k] f16 (LDS-tiled transpose)
// block (32,8), grid (T/32, K/32, B)
// ---------------------------------------------------------------------------
__global__ __launch_bounds__(256)
void transpose_h_kernel(const float* __restrict__ H, _Float16* __restrict__ Ht) {
    __shared__ float tile[32][33];
    const int b  = blockIdx.z;
    const int k0 = blockIdx.y * 32;
    const int t0 = blockIdx.x * 32;
    const int tx = threadIdx.x;   // 0..31
    const int ty = threadIdx.y;   // 0..7

    const float* src = H + (size_t)b * K_DIM * T_DIM;
#pragma unroll
    for (int i = 0; i < 4; ++i)
        tile[ty + 8*i][tx] = src[(size_t)(k0 + ty + 8*i) * T_DIM + t0 + tx];

    __syncthreads();

    _Float16* dst = Ht + (size_t)b * T_DIM * K_DIM;
#pragma unroll
    for (int i = 0; i < 4; ++i)
        dst[(size_t)(t0 + ty + 8*i) * K_DIM + k0 + tx] = (_Float16)tile[tx][ty + 8*i];
}

// ---------------------------------------------------------------------------
// Fused GEMM (WMMA f16 -> f32) + structured-latent epilogue
// grid (4 t-tiles, 16 atom-tiles, 64 batches), 256 threads = 8 waves.
// wave j -> 32 columns [tTile*256 + j*32, +32) as two 16-col sub-tiles;
// each A (W) fragment feeds two WMMAs. 5 row groups:
//   g=0: rows a0..a0+15            (raw_k)
//   g=1: rows 256+a0..+15          (raw_theta)
//   g=2..4: rows 512+3*a0 .. +47   (logits, 3 per atom)
// => 4 K-steps * 5 groups * 2 col-tiles = 40 v_wmma per wave.
// Epilogue: atom-major lanes, float4 LDS readback, float4 stores.
// ---------------------------------------------------------------------------
__device__ __forceinline__ float softplus_fast(float x) {
    // log1p(exp(x)) = max(x,0) + log(1 + exp(-|x|))
    return fmaxf(x, 0.f) + __logf(1.f + __expf(-fabsf(x)));
}

__global__ __launch_bounds__(256)
void fused_gemm_epilogue(const _Float16* __restrict__ Wh,
                         const _Float16* __restrict__ Ht,
                         const float*    __restrict__ bias,
                         float*          __restrict__ out) {
    __shared__ float lds[8][5][256];   // per-wave staging: 5 tiles of 16x16 f32

    const int wave = threadIdx.x >> 5;
    const int lane = threadIdx.x & 31;
    const int n    = lane & 15;        // column within tile / M row for A frags
    const int half = lane >> 4;        // K-half selector per ISA layout

    const int tTile = blockIdx.x;      // 0..3   (256 columns each)
    const int a0    = blockIdx.y << 4; // atom base (16 atoms per block)
    const int bIdx  = blockIdx.z;      // 0..63

    const int tcol = tTile * 256 + wave * 32 + n;   // first column sub-tile

    int rowg[5];
    rowg[0] = a0 + n;                  // raw_k rows
    rowg[1] = ATOMS + a0 + n;          // raw_theta rows
    rowg[2] = 2*ATOMS + 3*a0 + n;      // logit block rows 0..15
    rowg[3] = rowg[2] + 16;            // rows 16..31
    rowg[4] = rowg[2] + 32;            // rows 32..47

    // Epilogue lane mapping: one atom per lane, half a row of columns.
    const int ap   = lane >> 1;             // atom within tile: 0..15
    const int nh   = (lane & 1) * 8;        // column half: 0 or 8
    const int atom = a0 + ap;

    // Per-lane bias values (issued early; overlap with the GEMM below).
    const float bK  = bias[atom];
    const float bTh = bias[ATOMS + atom];
    const float bL0 = bias[2*ATOMS + 3*atom];
    const float bL1 = bias[2*ATOMS + 3*atom + 1];
    const float bL2 = bias[2*ATOMS + 3*atom + 2];

    const _Float16* hcol = Ht + ((size_t)bIdx * T_DIM + tcol) * K_DIM;

    v8f zero = {};
    v8f acc[5][2];
#pragma unroll
    for (int g = 0; g < 5; ++g) { acc[g][0] = zero; acc[g][1] = zero; }

#pragma unroll
    for (int s = 0; s < 4; ++s) {      // K = 4 * 32
        const int k0 = 32 * s;
        // Two B fragments (16 columns apart => +16*128 f16 elements)
        v8h b0lo = *(const v8h*)(hcol + k0 + 8*half);
        v8h b0hi = *(const v8h*)(hcol + k0 + 16 + 8*half);
        v8h b1lo = *(const v8h*)(hcol + 16*K_DIM + k0 + 8*half);
        v8h b1hi = *(const v8h*)(hcol + 16*K_DIM + k0 + 16 + 8*half);
        v16h bf0 = __builtin_shufflevector(b0lo, b0hi,
                     0,1,2,3,4,5,6,7,8,9,10,11,12,13,14,15);
        v16h bf1 = __builtin_shufflevector(b1lo, b1hi,
                     0,1,2,3,4,5,6,7,8,9,10,11,12,13,14,15);
#pragma unroll
        for (int g = 0; g < 5; ++g) {
            const _Float16* arow = Wh + (size_t)rowg[g] * K_DIM;
            v8h alo = *(const v8h*)(arow + k0 + 8*half);
            v8h ahi = *(const v8h*)(arow + k0 + 16 + 8*half);
            v16h af = __builtin_shufflevector(alo, ahi,
                        0,1,2,3,4,5,6,7,8,9,10,11,12,13,14,15);
            acc[g][0] = __builtin_amdgcn_wmma_f32_16x16x32_f16(
                            false, af, false, bf0, (short)0, acc[g][0], false, false);
            acc[g][1] = __builtin_amdgcn_wmma_f32_16x16x32_f16(
                            false, af, false, bf1, (short)0, acc[g][1], false, false);
        }
    }

    // Row pointers into this wave's LDS staging region (3 logit rows per atom).
    const int p0 = 3 * ap;
    const float* rowK  = &lds[wave][0][ap * 16 + nh];
    const float* rowTh = &lds[wave][1][ap * 16 + nh];
    const float* rowL0 = &lds[wave][2 + ((p0    ) >> 4)][((p0    ) & 15) * 16 + nh];
    const float* rowL1 = &lds[wave][2 + ((p0 + 1) >> 4)][((p0 + 1) & 15) * 16 + nh];
    const float* rowL2 = &lds[wave][2 + ((p0 + 2) >> 4)][((p0 + 2) & 15) * 16 + nh];

    // Epilogue: two column sub-tiles through the same 40KB LDS staging buffer.
#pragma unroll 1
    for (int c = 0; c < 2; ++c) {
        // Spill: element r of acc holds D[M = 8*half + r][N = n]
#pragma unroll
        for (int g = 0; g < 5; ++g)
#pragma unroll
            for (int r = 0; r < 8; ++r)
                lds[wave][g][(8*half + r) * 16 + n] = acc[g][c][r];

        __syncthreads();

        // Vector readback: 10 x ds_load_b128 per lane (conflict-free: 32B/lane).
        float kr[8], tr[8], la[8], lb[8], lc[8];
        *(float4*)&kr[0] = *(const float4*)&rowK [0];
        *(float4*)&kr[4] = *(const float4*)&rowK [4];
        *(float4*)&tr[0] = *(const float4*)&rowTh[0];
        *(float4*)&tr[4] = *(const float4*)&rowTh[4];
        *(float4*)&la[0] = *(const float4*)&rowL0[0];
        *(float4*)&la[4] = *(const float4*)&rowL0[4];
        *(float4*)&lb[0] = *(const float4*)&rowL1[0];
        *(float4*)&lb[4] = *(const float4*)&rowL1[4];
        *(float4*)&lc[0] = *(const float4*)&rowL2[0];
        *(float4*)&lc[4] = *(const float4*)&rowL2[4];

        float zo[8];
#pragma unroll
        for (int j = 0; j < 8; ++j) {
            float kraw  = kr[j] + bK;
            float thraw = tr[j] + bTh;
            float l0    = la[j] + bL0;
            float l1    = lb[j] + bL1;
            float l2    = lc[j] + bL2;

            float kk  = fminf(softplus_fast(kraw) + 0.1f, 1000.f);
            float th  = fminf(fmaxf(softplus_fast(thraw) + 1e-6f, 1e-4f), 100.f);
            float mag = kk * th;

            float mx  = fmaxf(l0, fmaxf(l1, l2));
            float e0  = __expf(l0 - mx);
            float e1  = __expf(l1 - mx);
            float e2  = __expf(l2 - mx);
            float inv = 1.f / (e0 + e1 + e2);
            float delta = (e2 - e0) * inv;
            float zp    = e1 * inv;                       // zero-probability
            zo[j] = (zp < 0.95f) ? mag * delta : 0.f;     // mask: zp < 1-EPS
        }

        const int tBase = tTile * 256 + wave * 32 + c * 16;
        float* optr = out + ((size_t)bIdx * ATOMS + atom) * T_DIM + tBase + nh;
        *(float4*)&optr[0] = *(const float4*)&zo[0];
        *(float4*)&optr[4] = *(const float4*)&zo[4];

        __syncthreads();
    }
}

// ---------------------------------------------------------------------------
extern "C" void kernel_launch(void* const* d_in, const int* in_sizes, int n_in,
                              void* d_out, int out_size, void* d_ws, size_t ws_size,
                              hipStream_t stream) {
    const float* h = (const float*)d_in[0];   // (64,128,1024) f32
    const float* W = (const float*)d_in[1];   // (1280,128)    f32
    const float* b = (const float*)d_in[2];   // (1280,)       f32
    float* out     = (float*)d_out;           // (64,256,1024) f32

    // Workspace layout: [W as f16: 1280*128*2 B][Ht as f16: 64*1024*128*2 B]
    _Float16* Wh = (_Float16*)d_ws;
    _Float16* Ht = (_Float16*)((char*)d_ws + (size_t)OUT_ROWS * K_DIM * sizeof(_Float16));

    convert_w_kernel<<<dim3((OUT_ROWS * K_DIM) / 256), dim3(256), 0, stream>>>(W, Wh);
    transpose_h_kernel<<<dim3(T_DIM / 32, K_DIM / 32, 64), dim3(32, 8), 0, stream>>>(h, Ht);
    fused_gemm_epilogue<<<dim3(4, 16, 64), dim3(256), 0, stream>>>(Wh, Ht, b, out);
}
